// VectorQuantizerEMA_2259152797971
// MI455X (gfx1250) — matrix operand order repaired and checked by reading, not compiled, not run on hardware
//
#include <hip/hip_runtime.h>
#include <hip/hip_bf16.h>

#define K_EMB 512
#define D_EMB 64
#define T_LEN 4096
#define BATCH 32
#define N_ROWS (BATCH * T_LEN)        // 131072
#define ROWS_PER_BLOCK 128
#define SH 72                         // half-element stride: 144 B = 36 banks -> conflict-free, 16B-aligned rows
#define SF 66                         // float stride for the f32 x tile

typedef __attribute__((ext_vector_type(4)))  float     v4f;
typedef __attribute__((ext_vector_type(8)))  float     v8f;
typedef __attribute__((ext_vector_type(8)))  _Float16  v8h;
typedef __attribute__((ext_vector_type(16))) _Float16  v16h;

// Assemble a v16h WMMA fragment from two 16-byte LDS chunks.
__device__ __forceinline__ v16h ld_frag(const _Float16* p0, const _Float16* p1) {
  v8h a = *(const v8h*)p0;
  v8h b = *(const v8h*)p1;
  v16h r;
  #pragma unroll
  for (int i = 0; i < 8; ++i) { r[i] = a[i]; r[i + 8] = b[i]; }
  return r;
}

// ---------------------------------------------------------------------------
// Kernel 1: zero scratch (counts, dw) and the loss accumulator.
// ---------------------------------------------------------------------------
__global__ void vq_init(float* __restrict__ counts, float* __restrict__ dw,
                        float* __restrict__ o_loss) {
  int g = blockIdx.x * blockDim.x + threadIdx.x;
  if (g < K_EMB) counts[g] = 0.0f;
  if (g < K_EMB * D_EMB) dw[g] = 0.0f;
  if (g == 0) o_loss[0] = 0.0f;
}

// ---------------------------------------------------------------------------
// Kernel 2: distances via error-compensated f16 WMMA (f16x3 split), argmin,
// quantized gather, EMA stats. 256 thr = 8 waves, 16 rows/wave, 128 rows/block.
// ---------------------------------------------------------------------------
__global__ __launch_bounds__(256, 1) void vq_main(
    const float* __restrict__ x,      // [32][64][4096]
    const float* __restrict__ emb,    // [512][64]
    float* __restrict__ o_loss,       // [1]
    float* __restrict__ o_q,          // [32][64][4096]
    float* __restrict__ o_idx,        // [32][4096] (stored as float)
    float* __restrict__ counts,       // [512] scratch
    float* __restrict__ dw)           // [512*64] scratch
{
  __shared__ __align__(16) _Float16 es_h[K_EMB * SH];           // codebook hi
  __shared__ __align__(16) _Float16 es_l[K_EMB * SH];           // codebook lo (residual)
  __shared__ __align__(16) _Float16 xs_h[ROWS_PER_BLOCK * SH];  // x hi
  __shared__ __align__(16) _Float16 xs_l[ROWS_PER_BLOCK * SH];  // x lo
  __shared__ float xs_f[ROWS_PER_BLOCK * SF];                   // x f32 (for dw / loss)
  __shared__ float eN[K_EMB];                                   // ||eh+el||^2
  __shared__ int   ridx[ROWS_PER_BLOCK];

  const int tid     = threadIdx.x;
  const int rowbase = blockIdx.x * ROWS_PER_BLOCK;
  const int b       = rowbase / T_LEN;            // 128 | 4096 -> one b per block
  const int tbase   = rowbase % T_LEN;

  // ---- Phase 1: stage codebook (hi/lo f16) + x tile (hi/lo f16 + f32) ----
  #pragma unroll 4
  for (int i = 0; i < 32; ++i) {                  // 8192 float4 chunks of emb
    int lin = tid + i * 256;
    int k   = lin >> 4;
    int d4  = (lin & 15) << 2;
    v4f v = *(const v4f*)(emb + k * D_EMB + d4);
    #pragma unroll
    for (int j = 0; j < 4; ++j) {
      float f = v[j];
      _Float16 h = (_Float16)f;
      es_h[k * SH + d4 + j] = h;
      es_l[k * SH + d4 + j] = (_Float16)(f - (float)h);
    }
  }
  #pragma unroll 4
  for (int i = 0; i < 32; ++i) {                  // x[b][d][tbase+r], coalesced over r
    int lin = tid + i * 256;
    int d   = lin >> 7;
    int r   = lin & 127;
    float f = x[(size_t)(b * D_EMB + d) * T_LEN + tbase + r];
    xs_f[r * SF + d] = f;
    _Float16 h = (_Float16)f;
    xs_h[r * SH + d] = h;
    xs_l[r * SH + d] = (_Float16)(f - (float)h);
  }
  __syncthreads();
  // ||e||^2 from the SAME hi+lo representation used by the matmul.
  #pragma unroll
  for (int i = 0; i < 2; ++i) {
    int k = tid + i * 256;
    float s = 0.0f;
    #pragma unroll
    for (int d = 0; d < D_EMB; ++d) {
      float e = (float)es_h[k * SH + d] + (float)es_l[k * SH + d];
      s += e * e;
    }
    eN[k] = s;
  }
  __syncthreads();

  // ---- Phase 2: f16x3 WMMA distance tiles + running argmin ----
  const int wave = tid >> 5;
  const int lane = tid & 31;
  const int col  = lane & 15;   // C-layout: N = lane & 15
  const int hi   = lane >> 4;   // C rows 0-7 (lanes 0-15) / 8-15 (lanes 16-31)
  const int wrow = wave * 16;

  // A frags (16x32 f16): lane half selects K-subblock: chunks at d=32w+8*hi, +16.
  v16h ah[2], al[2];
  {
    const _Float16* rh = xs_h + (wrow + col) * SH + 8 * hi;
    const _Float16* rl = xs_l + (wrow + col) * SH + 8 * hi;
    #pragma unroll
    for (int w = 0; w < 2; ++w) {
      ah[w] = ld_frag(rh + 32 * w, rh + 32 * w + 16);
      al[w] = ld_frag(rl + 32 * w, rl + 32 * w + 16);
    }
  }

  float minv[8];
  int   mini[8];
  #pragma unroll
  for (int r = 0; r < 8; ++r) { minv[r] = 3.4e38f; mini[r] = 0; }

  for (int kt = 0; kt < K_EMB / 16; ++kt) {
    const int kb = kt * 16;
    // B frags (32x16 f16): lane n holds K chunks at d=32w+8*hi, +16 (matches A).
    const _Float16* eh = es_h + (kb + col) * SH + 8 * hi;
    const _Float16* el = es_l + (kb + col) * SH + 8 * hi;
    v8f c = {0.f, 0.f, 0.f, 0.f, 0.f, 0.f, 0.f, 0.f};
    #pragma unroll
    for (int w = 0; w < 2; ++w) {
      v16h bh = ld_frag(eh + 32 * w, eh + 32 * w + 16);
      v16h bl = ld_frag(el + 32 * w, el + 32 * w + 16);
      // x.e ~= xh.eh + xl.eh + xh.el  (f32 accumulate)
      c = __builtin_amdgcn_wmma_f32_16x16x32_f16(false, ah[w], false, bh,
                                                 (short)0, c, false, false);
      c = __builtin_amdgcn_wmma_f32_16x16x32_f16(false, al[w], false, bh,
                                                 (short)0, c, false, false);
      c = __builtin_amdgcn_wmma_f32_16x16x32_f16(false, ah[w], false, bl,
                                                 (short)0, c, false, false);
    }
    // dist(m, kb+col) = ||e||^2 - 2*dot   (||x||^2 constant per row: dropped)
    const float en = eN[kb + col];
    #pragma unroll
    for (int r = 0; r < 8; ++r) {
      float dist = en - 2.0f * c[r];
      if (dist < minv[r]) { minv[r] = dist; mini[r] = kb + col; }
    }
  }

  // Argmin reduce across the 16 lanes per row (wave32 halves are independent).
  #pragma unroll
  for (int r = 0; r < 8; ++r) {
    float v  = minv[r];
    int   id = mini[r];
    #pragma unroll
    for (int off = 1; off <= 8; off <<= 1) {
      float ov = __shfl_xor(v, off, 32);
      int   oi = __shfl_xor(id, off, 32);
      if (ov < v || (ov == v && oi < id)) { v = ov; id = oi; }
    }
    if (col == 0) ridx[wrow + hi * 8 + r] = id;   // lanes 0 and 16 write
  }
  __syncthreads();

  // ---- Phase 3: outputs + EMA stats + commitment loss ----
  if (tid < ROWS_PER_BLOCK) {
    int id = ridx[tid];
    o_idx[rowbase + tid] = (float)id;
    atomicAdd(&counts[id], 1.0f);
  }

  float lossAcc = 0.0f;
  #pragma unroll 4
  for (int i = 0; i < 32; ++i) {
    int lin = tid + i * 256;
    int d   = lin >> 7;
    int r   = lin & 127;
    int id  = ridx[r];
    float q  = emb[id * D_EMB + d];     // 128 KB table: L2/WGP$-resident gather
    float xv = xs_f[r * SF + d];
    o_q[(size_t)(b * D_EMB + d) * T_LEN + tbase + r] = q; // quantized_st == quantized
    atomicAdd(&dw[id * D_EMB + d], xv);
    float diff = q - xv;
    lossAcc += diff * diff;
  }
  #pragma unroll
  for (int off = 16; off >= 1; off >>= 1)
    lossAcc += __shfl_xor(lossAcc, off, 32);
  if (lane == 0)
    atomicAdd(o_loss, lossAcc * (0.25f / ((float)N_ROWS * (float)D_EMB)));
}

// ---------------------------------------------------------------------------
// Kernel 3: EMA finalize (1 block x 512 threads, one code per thread).
// ---------------------------------------------------------------------------
__global__ __launch_bounds__(512, 1) void vq_finalize(
    const float* __restrict__ ema_cs, const float* __restrict__ ema_w,
    const float* __restrict__ counts, const float* __restrict__ dw,
    float* __restrict__ o_emb, float* __restrict__ o_cs,
    float* __restrict__ o_emaw)
{
  __shared__ float red[K_EMB];
  const int k = threadIdx.x;
  float ncs = 0.99f * ema_cs[k] + 0.01f * counts[k];
  o_cs[k] = ncs;
  red[k] = ncs;
  __syncthreads();
  for (int s = 256; s > 0; s >>= 1) {
    if (k < s) red[k] += red[k + s];
    __syncthreads();
  }
  const float n     = red[0];
  const float csize = (ncs + 1e-5f) / (n + (float)K_EMB * 1e-5f) * n;
  const float inv   = 1.0f / csize;
  #pragma unroll 8
  for (int d = 0; d < D_EMB; ++d) {
    float nw = 0.99f * ema_w[k * D_EMB + d] + 0.01f * dw[k * D_EMB + d];
    o_emaw[k * D_EMB + d] = nw;
    o_emb[k * D_EMB + d]  = nw * inv;
  }
}

// ---------------------------------------------------------------------------
extern "C" void kernel_launch(void* const* d_in, const int* in_sizes, int n_in,
                              void* d_out, int out_size, void* d_ws, size_t ws_size,
                              hipStream_t stream) {
  const float* x      = (const float*)d_in[0];  // [32,64,4096]
  const float* emb    = (const float*)d_in[1];  // [512,64]
  const float* ema_cs = (const float*)d_in[2];  // [512]
  const float* ema_w  = (const float*)d_in[3];  // [512,64]

  float* out    = (float*)d_out;                // tuple, flat in return order
  float* o_loss = out;
  float* o_q    = out + 1;
  float* o_idx  = o_q   + (size_t)BATCH * D_EMB * T_LEN;
  float* o_emb  = o_idx + (size_t)BATCH * T_LEN;
  float* o_cs   = o_emb + (size_t)K_EMB * D_EMB;
  float* o_emaw = o_cs  + K_EMB;

  float* counts = (float*)d_ws;                 // [512]
  float* dw     = counts + K_EMB;               // [512*64]

  vq_init<<<(K_EMB * D_EMB + 255) / 256, 256, 0, stream>>>(counts, dw, o_loss);
  vq_main<<<N_ROWS / ROWS_PER_BLOCK, 256, 0, stream>>>(
      x, emb, o_loss, o_q, o_idx, counts, dw);
  vq_finalize<<<1, K_EMB, 0, stream>>>(ema_cs, ema_w, counts, dw,
                                       o_emb, o_cs, o_emaw);
}
